// MDN_RNN_55327768708624
// MI455X (gfx1250) — matrix operand order repaired and checked by reading, not compiled, not run on hardware
//
#include <hip/hip_runtime.h>
#include <hip/hip_bf16.h>
#include <math.h>

// ---------------------------------------------------------------------------
// MDN-RNN fused step for gfx1250 (MI455X), wave32 + v_wmma_f32_16x16x32_bf16
// M-tile: 32 rows per wave (2 strips of 16), 64 rows per 2-wave block.
// ---------------------------------------------------------------------------

typedef __attribute__((ext_vector_type(16))) __bf16 v16bf;
typedef __attribute__((ext_vector_type(8)))  float  v8f;

union alignas(32) Frag {
    v16bf v;
    unsigned int u[8];
};

#define B_INPUT   35
#define B_HIDDEN  256
#define KPAD1     320          // 35 + 256 padded to multiple of 32
#define NKT1      10           // 320/32 K-steps for gates GEMM
#define NGATE_NT  64           // 1024/16 N tiles (i,f,g,o offset by 256)
#define NKT2      8            // 256/32 K-steps for GMM GEMM
#define NPAD2     336          // 325 padded to multiple of 16
#define NT2       21           // 336/16 N tiles
#define LOG_TEMP  0.13976194f  // ln(1.15)

#define MROWS     64           // rows per block (2 waves x 32 rows)
#define SY_BYTES  (MROWS * NPAD2 * 4)      // 86016  (also covers sA: 40960)
#define SH_BYTES  (MROWS * 256 * 2)        // 32768

// ws layout (bytes)
#define WP1_U32   (NKT1 * NGATE_NT * 32 * 8)     // 163840 dwords
#define WP2_U32   (NKT2 * NT2 * 32 * 8)          // 43008 dwords
#define WP1_OFF   0
#define WP2_OFF   (WP1_U32 * 4)                  // 655360
#define B1_OFF    (WP2_OFF + WP2_U32 * 4)        // +172032
#define B2_OFF    (B1_OFF + 1024 * 4)

__device__ __forceinline__ unsigned short f2bf(float f) {
    unsigned int u = __float_as_uint(f);
    u += 0x7FFFu + ((u >> 16) & 1u);             // round to nearest even
    return (unsigned short)(u >> 16);
}
__device__ __forceinline__ unsigned int pack2bf(float lo, float hi) {
    return (unsigned int)f2bf(lo) | ((unsigned int)f2bf(hi) << 16);
}
__device__ __forceinline__ float sigm(float x) {
    return 1.0f / (1.0f + __expf(-x));
}
__device__ __forceinline__ unsigned int hashu(unsigned int x) {
    x ^= x >> 16; x *= 0x7feb352dU;
    x ^= x >> 15; x *= 0x846ca68bU;
    x ^= x >> 16; return x;
}
__device__ __forceinline__ float u01(unsigned int h) {
    return ((float)(h >> 8) + 0.5f) * (1.0f / 16777216.0f);
}
__device__ __forceinline__ float gate_w(const float* __restrict__ W_ih,
                                        const float* __restrict__ W_hh,
                                        int n, int K) {
    if (K < B_INPUT)             return W_ih[n * B_INPUT + K];
    if (K < B_INPUT + B_HIDDEN)  return W_hh[n * B_HIDDEN + (K - B_INPUT)];
    return 0.0f;
}

// ---------------------------------------------------------------------------
// Kernel 0: pack weights into WMMA B-fragment layout (bf16, contiguous K order), fuse biases.
// B fragment (16-bit, 32x16): lane l -> n = nt*16 + (l&15),
//   K = kt*32 + (l>>4)*16 + 2*e (+0,+1 packed in one dword), e = dword 0..7.
// ---------------------------------------------------------------------------
__global__ void pack_weights(const float* __restrict__ W_ih,
                             const float* __restrict__ W_hh,
                             const float* __restrict__ b_ih,
                             const float* __restrict__ b_hh,
                             const float* __restrict__ W_gmm,
                             const float* __restrict__ b_gmm,
                             unsigned int* __restrict__ wp1,
                             unsigned int* __restrict__ wp2,
                             float* __restrict__ bias1,
                             float* __restrict__ bias2) {
    int idx = blockIdx.x * 256 + threadIdx.x;
    if (idx < WP1_U32) {
        int e    = idx & 7;
        int lane = (idx >> 3) & 31;
        int frag = idx >> 8;                 // kt*64 + nt
        int nt   = frag & 63;
        int kt   = frag >> 6;
        int n    = nt * 16 + (lane & 15);
        int K    = kt * 32 + ((lane >> 4) * 16) + 2 * e;
        wp1[idx] = pack2bf(gate_w(W_ih, W_hh, n, K),
                           gate_w(W_ih, W_hh, n, K + 1));
    } else if (idx < WP1_U32 + WP2_U32) {
        int t    = idx - WP1_U32;
        int e    = t & 7;
        int lane = (t >> 3) & 31;
        int frag = t >> 8;                   // kt*21 + nt
        int nt   = frag % NT2;
        int kt   = frag / NT2;
        int n    = nt * 16 + (lane & 15);
        int K    = kt * 32 + ((lane >> 4) * 16) + 2 * e;
        float lo = (n < 325) ? W_gmm[n * 256 + K]     : 0.0f;
        float hi = (n < 325) ? W_gmm[n * 256 + K + 1] : 0.0f;
        wp2[t] = pack2bf(lo, hi);
    } else if (idx < WP1_U32 + WP2_U32 + 1024) {
        int j = idx - (WP1_U32 + WP2_U32);
        bias1[j] = b_ih[j] + b_hh[j];
    } else if (idx < WP1_U32 + WP2_U32 + 1024 + NPAD2) {
        int j = idx - (WP1_U32 + WP2_U32 + 1024);
        bias2[j] = (j < 325) ? b_gmm[j] : 0.0f;
    }
}

// ---------------------------------------------------------------------------
// Kernel 1: fused LSTM cell + GMM head + sampling.
// 64 threads = 2 waves; wave owns 32 rows (2 WMMA strips); block = 64 rows.
// ---------------------------------------------------------------------------
__global__ __launch_bounds__(64) void mdnrnn_fused(
        const float* __restrict__ x,
        const float* __restrict__ h,
        const float* __restrict__ c,
        const unsigned int* __restrict__ wp1,
        const unsigned int* __restrict__ wp2,
        const float* __restrict__ bias1,
        const float* __restrict__ bias2,
        float* __restrict__ out) {
    // LDS: region A (86016B) = sA (x|h bf16, phase 1) then sY (y f32, phase 3)
    //      region B (32768B) = sH (h_new bf16)
    __shared__ unsigned char smem[SY_BYTES + SH_BYTES];
    unsigned short* sA = (unsigned short*)smem;
    float*          sY = (float*)smem;
    unsigned short* sH = (unsigned short*)(smem + SY_BYTES);

    const int tid     = threadIdx.x;
    const int w       = tid >> 5;
    const int lane    = tid & 31;
    const int m       = lane & 15;          // row within strip (A) / col (B,C)
    const int hi16    = lane >> 4;          // lane half
    const int kb      = hi16 * 16;          // K sub-offset for A fragments
    const int rowBase = blockIdx.x * MROWS;
    const int wrow    = w * 32;             // wave's first local row

    // ---- stage x|h as bf16 into sA, K padded to 320 -----------------------
    for (int i = tid; i < MROWS * KPAD1; i += 64) {
        int r = i / KPAD1, k = i - r * KPAD1;
        int g = rowBase + r;
        float v = (k < B_INPUT) ? x[g * B_INPUT + k]
                : (k < B_INPUT + B_HIDDEN) ? h[g * B_HIDDEN + (k - B_INPUT)]
                : 0.0f;
        sA[i] = f2bf(v);
    }
    __syncthreads();

    // ---- gates GEMM + LSTM nonlinearity -----------------------------------
    // A fragments streamed from LDS per K-step (16-bit 16x32 layout):
    //   lane -> row m, dwords 0..7 at K=kt*32+kb.. (contiguous 16).
    for (int jt = 0; jt < 16; ++jt) {
        v8f ai0 = {}, af0 = {}, ag0 = {}, ao0 = {};
        v8f ai1 = {}, af1 = {}, ag1 = {}, ao1 = {};
#pragma unroll
        for (int kt = 0; kt < NKT1; ++kt) {
            Frag a0, a1;
            const uint4* p0 = (const uint4*)&sA[(wrow + m) * KPAD1 + kt * 32 + kb];
            const uint4* p1 = (const uint4*)&sA[(wrow + 16 + m) * KPAD1 + kt * 32 + kb];
            ((uint4*)a0.u)[0] = p0[0]; ((uint4*)a0.u)[1] = p0[1];
            ((uint4*)a1.u)[0] = p1[0]; ((uint4*)a1.u)[1] = p1[1];
            Frag b0, b1, b2, b3;
            const uint4* q;
            q = (const uint4*)&wp1[((kt * NGATE_NT + jt     ) * 32 + lane) * 8];
            ((uint4*)b0.u)[0] = q[0]; ((uint4*)b0.u)[1] = q[1];
            q = (const uint4*)&wp1[((kt * NGATE_NT + jt + 16) * 32 + lane) * 8];
            ((uint4*)b1.u)[0] = q[0]; ((uint4*)b1.u)[1] = q[1];
            q = (const uint4*)&wp1[((kt * NGATE_NT + jt + 32) * 32 + lane) * 8];
            ((uint4*)b2.u)[0] = q[0]; ((uint4*)b2.u)[1] = q[1];
            q = (const uint4*)&wp1[((kt * NGATE_NT + jt + 48) * 32 + lane) * 8];
            ((uint4*)b3.u)[0] = q[0]; ((uint4*)b3.u)[1] = q[1];
            ai0 = __builtin_amdgcn_wmma_f32_16x16x32_bf16(false, a0.v, false, b0.v, (short)0, ai0, false, false);
            af0 = __builtin_amdgcn_wmma_f32_16x16x32_bf16(false, a0.v, false, b1.v, (short)0, af0, false, false);
            ag0 = __builtin_amdgcn_wmma_f32_16x16x32_bf16(false, a0.v, false, b2.v, (short)0, ag0, false, false);
            ao0 = __builtin_amdgcn_wmma_f32_16x16x32_bf16(false, a0.v, false, b3.v, (short)0, ao0, false, false);
            ai1 = __builtin_amdgcn_wmma_f32_16x16x32_bf16(false, a1.v, false, b0.v, (short)0, ai1, false, false);
            af1 = __builtin_amdgcn_wmma_f32_16x16x32_bf16(false, a1.v, false, b1.v, (short)0, af1, false, false);
            ag1 = __builtin_amdgcn_wmma_f32_16x16x32_bf16(false, a1.v, false, b2.v, (short)0, ag1, false, false);
            ao1 = __builtin_amdgcn_wmma_f32_16x16x32_bf16(false, a1.v, false, b3.v, (short)0, ao1, false, false);
        }
        const int col = jt * 16 + m;
        const float bi  = bias1[col];
        const float bff = bias1[col + 256];
        const float bg  = bias1[col + 512];
        const float bo  = bias1[col + 768];
#pragma unroll
        for (int s = 0; s < 2; ++s) {
            const v8f& ai = s ? ai1 : ai0;
            const v8f& af = s ? af1 : af0;
            const v8f& ag = s ? ag1 : ag0;
            const v8f& ao = s ? ao1 : ao0;
#pragma unroll
            for (int r = 0; r < 8; ++r) {
                int lrow = wrow + s * 16 + r + (hi16 << 3); // VGPR r -> M=r / r+8
                int grow = rowBase + lrow;
                float iv = sigm(ai[r] + bi);
                float fv = sigm(af[r] + bff);
                float gv = tanhf(ag[r] + bg);
                float ov = sigm(ao[r] + bo);
                float cn = fv * c[grow * 256 + col] + iv * gv;
                float hn = ov * tanhf(cn);
                sH[lrow * 256 + col] = f2bf(hn);
            }
        }
    }

    __syncthreads();   // sA dead everywhere; region reused as sY below

    // ---- GMM head GEMM, stage y into LDS ----------------------------------
    for (int nt = 0; nt < NT2; ++nt) {
        v8f acc0 = {}, acc1 = {};
#pragma unroll
        for (int kt = 0; kt < NKT2; ++kt) {
            Frag a0, a1, b;
            const uint4* p0 = (const uint4*)&sH[(wrow + m) * 256 + kt * 32 + kb];
            const uint4* p1 = (const uint4*)&sH[(wrow + 16 + m) * 256 + kt * 32 + kb];
            ((uint4*)a0.u)[0] = p0[0]; ((uint4*)a0.u)[1] = p0[1];
            ((uint4*)a1.u)[0] = p1[0]; ((uint4*)a1.u)[1] = p1[1];
            const uint4* q = (const uint4*)&wp2[((kt * NT2 + nt) * 32 + lane) * 8];
            ((uint4*)b.u)[0] = q[0]; ((uint4*)b.u)[1] = q[1];
            acc0 = __builtin_amdgcn_wmma_f32_16x16x32_bf16(false, a0.v, false, b.v, (short)0, acc0, false, false);
            acc1 = __builtin_amdgcn_wmma_f32_16x16x32_bf16(false, a1.v, false, b.v, (short)0, acc1, false, false);
        }
        const float bb = bias2[nt * 16 + m];
#pragma unroll
        for (int r = 0; r < 8; ++r) {
            int l0 = wrow + r + (hi16 << 3);
            int l1 = l0 + 16;
            sY[l0 * NPAD2 + nt * 16 + m] = acc0[r] + bb;
            sY[l1 * NPAD2 + nt * 16 + m] = acc1[r] + bb;
        }
    }

    // ---- per-row softmax + categorical + gaussian sampling ----------------
    // y cols: mixture mx at [mx*65] = pi logit, +1..32 = mus, +33..64 = logvars
    for (int rr = 0; rr < 32; ++rr) {
        int lrow = wrow + rr;
        int grow = rowBase + lrow;
        const float* yr = &sY[lrow * NPAD2];
        float p0 = yr[0], p1 = yr[65], p2 = yr[130], p3 = yr[195], p4 = yr[260];
        float mx = fmaxf(fmaxf(fmaxf(p0, p1), fmaxf(p2, p3)), p4);
        float e0 = __expf(p0 - mx), e1 = __expf(p1 - mx), e2 = __expf(p2 - mx);
        float e3 = __expf(p3 - mx), e4 = __expf(p4 - mx);
        float s = e0 + e1 + e2 + e3 + e4;
        float u = u01(hashu((unsigned)grow * 2654435761u ^ 0x9E3779B9u)) * s;
        int   idx = 0;
        float ca  = e0;
        if (u > ca) { idx = 1; ca += e1; }
        if (u > ca) { idx = 2; ca += e2; }
        if (u > ca) { idx = 3; ca += e3; }
        if (u > ca) { idx = 4; }
        // 32 lanes = 32 latent dims, coalesced store
        int j = lane;
        float mu = yr[idx * 65 + 1 + j];
        float lv = yr[idx * 65 + 33 + j];
        unsigned int h1 = hashu((unsigned)(grow * 32 + j) + 1u);
        unsigned int h2 = hashu((unsigned)(grow * 32 + j) ^ 0x85EBCA6Bu);
        float eps = sqrtf(-2.0f * __logf(u01(h1))) * __cosf(6.2831853f * u01(h2));
        out[grow * 32 + j] = mu + __expf(0.5f * (lv + LOG_TEMP)) * eps;
    }
}

// ---------------------------------------------------------------------------
extern "C" void kernel_launch(void* const* d_in, const int* in_sizes, int n_in,
                              void* d_out, int out_size, void* d_ws, size_t ws_size,
                              hipStream_t stream) {
    const float* x     = (const float*)d_in[0];
    const float* h     = (const float*)d_in[1];
    const float* c     = (const float*)d_in[2];
    const float* W_ih  = (const float*)d_in[3];
    const float* W_hh  = (const float*)d_in[4];
    const float* b_ih  = (const float*)d_in[5];
    const float* b_hh  = (const float*)d_in[6];
    const float* W_gmm = (const float*)d_in[7];
    const float* b_gmm = (const float*)d_in[8];

    const int B = in_sizes[0] / B_INPUT;   // 131072

    unsigned char* ws = (unsigned char*)d_ws;
    unsigned int* wp1 = (unsigned int*)(ws + WP1_OFF);
    unsigned int* wp2 = (unsigned int*)(ws + WP2_OFF);
    float* bias1      = (float*)(ws + B1_OFF);
    float* bias2      = (float*)(ws + B2_OFF);

    const int packN = WP1_U32 + WP2_U32 + 1024 + NPAD2;
    pack_weights<<<(packN + 255) / 256, 256, 0, stream>>>(
        W_ih, W_hh, b_ih, b_hh, W_gmm, b_gmm, wp1, wp2, bias1, bias2);

    mdnrnn_fused<<<B / MROWS, 64, 0, stream>>>(
        x, h, c, wp1, wp2, bias1, bias2, (float*)d_out);
}
